// EncodingLayer_17274358464618
// MI455X (gfx1250) — compile-verified, matrix operand order checked by reading, not run on hardware
//
#include <hip/hip_runtime.h>
#include <math.h>

// Problem constants (from reference)
#define B_ 4
#define S_ 2048
#define D_ 512
#define H_ 8
#define HD_ 64
#define DFF_ 2048
#define BS_ (B_ * S_)      // 8192
#define LN_EPS 1e-5f

typedef __attribute__((ext_vector_type(16))) __bf16 v16bf;
typedef __attribute__((ext_vector_type(8)))  __bf16 v8bf;
typedef __attribute__((ext_vector_type(8)))  float  v8f;

// ---------- scalar helpers ----------

__device__ __forceinline__ __bf16 f2bf(float f) {
  union { float f; unsigned u; } x; x.f = f;
  unsigned r = x.u + 0x7fffu + ((x.u >> 16) & 1u);   // round-to-nearest-even
  unsigned short h = (unsigned short)(r >> 16);
  union { unsigned short s; __bf16 b; } y; y.s = h;
  return y.b;
}

// ---------- CDNA5 async global->LDS copy (ASYNCcnt tracked) ----------

__device__ __forceinline__ void async_b128(unsigned lds_byte_off, const void* gptr) {
  unsigned long long g = (unsigned long long)gptr;
  asm volatile("global_load_async_to_lds_b128 %0, %1, off"
               :: "v"(lds_byte_off), "v"(g) : "memory");
}

template <int N> __device__ __forceinline__ void wait_async();
template <> __device__ __forceinline__ void wait_async<0>() {
  asm volatile("s_wait_asynccnt 0x0" ::: "memory");
}
template <> __device__ __forceinline__ void wait_async<4>() {
  asm volatile("s_wait_asynccnt 0x4" ::: "memory");
}
template <> __device__ __forceinline__ void wait_async<5>() {
  asm volatile("s_wait_asynccnt 0x5" ::: "memory");
}

// ---------- WMMA fragment helpers (ISA 7.12.2 16-bit layouts) ----------

__device__ __forceinline__ v16bf lds_frag_bf16(const __bf16* __restrict__ tile, int ld,
                                               int row0, int k0, int lane) {
  const int half = lane >> 4;
  const int r    = lane & 15;
  const __bf16* p = tile + (row0 + r) * ld + (k0 + half * 8);
  union { v16bf v; v8bf h[2]; } u;
  u.h[0] = *(const v8bf*)(p);
  u.h[1] = *(const v8bf*)(p + 16);
  return u.v;
}

__device__ __forceinline__ v16bf lds_frag_f32(const float* __restrict__ tile, int ld,
                                              int row0, int k0, int lane) {
  const int half = lane >> 4;
  const int r    = lane & 15;
  const float* p = tile + (row0 + r) * ld + (k0 + half * 8);
  v8f lo = *(const v8f*)(p);
  v8f hi = *(const v8f*)(p + 16);
  v16bf f;
#pragma unroll
  for (int i = 0; i < 8; ++i) { f[i] = f2bf(lo[i]); f[i + 8] = f2bf(hi[i]); }
  return f;
}

__device__ __forceinline__ v8f wmma_bf16(v16bf a, v16bf b, v8f c) {
  return __builtin_amdgcn_wmma_f32_16x16x32_bf16(false, a, false, b, (short)0, c,
                                                 false, false);
}

// ================= 128x128 block core (bf16 A), wave tile 32x64 =================
// 256 threads = 8 waves in a 4(M) x 2(N) grid. KS=32 double-buffered stages filled
// by async global->LDS DMA. Per k-step: 12 ds_load_b128 -> 8 back-to-back WMMAs.

template <int KS>
__device__ __forceinline__ void issue_tile128(const __bf16* __restrict__ src, int ld,
                                              int row0, int k0, unsigned so) {
  constexpr int CPR = KS / 8;        // 16B chunks per row
  constexpr int KSP = KS + 8;        // padded row stride (elements)
  constexpr int CNT = 128 * CPR;
  const int t = threadIdx.x;
#pragma unroll
  for (int c0 = 0; c0 < CNT; c0 += 256) {
    const int c   = c0 + t;
    const int row = c / CPR;
    const int cc  = c - row * CPR;
    async_b128(so + (unsigned)(row * KSP * 2 + cc * 16),
               src + (size_t)(row0 + row) * ld + (k0 + cc * 8));
  }
}

template <int KS>
__device__ __forceinline__ void gemm128x128(const __bf16* __restrict__ A, int lda,
                                            const __bf16* __restrict__ Bt, int ldb,
                                            int m_block, int n_block, int K,
                                            v8f acc[2][4], char* smem) {
  constexpr int KSP     = KS + 8;
  constexpr int T_BYTES = 128 * KSP * 2;                   // one tile stage
  constexpr int OPS     = 2 * ((128 * (KS / 8)) / 256);    // async ops/thread/stage
  static_assert(OPS == 4, "expected 4 ops per stage");
  __bf16* sA = (__bf16*)smem;                              // [2][128][KSP]
  __bf16* sB = (__bf16*)(smem + 2 * T_BYTES);              // [2][128][KSP]
  const unsigned sAo = (unsigned)(size_t)(void*)sA;
  const unsigned sBo = (unsigned)(size_t)(void*)sB;
  const int wid = threadIdx.x >> 5, lane = threadIdx.x & 31;
  const int widM = wid & 3, widN = wid >> 2;
  const int nst = K / KS;

  issue_tile128<KS>(A,  lda, m_block, 0, sAo);
  issue_tile128<KS>(Bt, ldb, n_block, 0, sBo);
  for (int s = 0; s < nst; ++s) {
    const int cur = s & 1;
    if (s + 1 < nst) {
      const int nxt = (s + 1) & 1;
      issue_tile128<KS>(A,  lda, m_block, (s + 1) * KS, sAo + (unsigned)(nxt * T_BYTES));
      issue_tile128<KS>(Bt, ldb, n_block, (s + 1) * KS, sBo + (unsigned)(nxt * T_BYTES));
      wait_async<OPS>();            // stage-s done, stage-(s+1) still in flight
    } else {
      wait_async<0>();
    }
    __syncthreads();
    const __bf16* tA = sA + (size_t)cur * (T_BYTES / 2);
    const __bf16* tB = sB + (size_t)cur * (T_BYTES / 2);
#pragma unroll
    for (int kk = 0; kk < KS; kk += 32) {
      v16bf a0 = lds_frag_bf16(tA, KSP, widM * 32,      kk, lane);
      v16bf a1 = lds_frag_bf16(tA, KSP, widM * 32 + 16, kk, lane);
      v16bf bfr[4];
#pragma unroll
      for (int j = 0; j < 4; ++j)
        bfr[j] = lds_frag_bf16(tB, KSP, widN * 64 + j * 16, kk, lane);
#pragma unroll
      for (int j = 0; j < 4; ++j) acc[0][j] = wmma_bf16(a0, bfr[j], acc[0][j]);
#pragma unroll
      for (int j = 0; j < 4; ++j) acc[1][j] = wmma_bf16(a1, bfr[j], acc[1][j]);
    }
    __syncthreads();                // protect buffer before stage s+2 overwrites
  }
}

#define SMEM128(KS) (4 * 128 * ((KS) + 8) * 2)

__device__ __forceinline__ void zero_acc24(v8f acc[2][4]) {
#pragma unroll
  for (int mi = 0; mi < 2; ++mi)
#pragma unroll
    for (int j = 0; j < 4; ++j)
#pragma unroll
      for (int i = 0; i < 8; ++i) acc[mi][j][i] = 0.0f;
}

// ================= 128x64 block core (f32 A) for atn @ V =================
// 8 waves stacked along M, wave tile 16x64, KS=32, A staged as f32 in LDS.

template <typename AT, int KS>
struct StageCfg {
  static constexpr int EPCA  = 16 / (int)sizeof(AT);
  static constexpr int KSPA  = KS + EPCA;
  static constexpr int KSPB  = KS + 8;
  static constexpr int CPRA  = KS / EPCA;
  static constexpr int CPRB  = KS / 8;
  static constexpr int CA    = 128 * CPRA;
  static constexpr int CB    = 64 * CPRB;
  static constexpr int A_BYTES = 128 * KSPA * (int)sizeof(AT);
  static constexpr int B_BYTES = 64 * KSPB * 2;
  static constexpr int OPS   = CA / 256 + CB / 256;
  static constexpr int SMEM  = 2 * (A_BYTES + B_BYTES);
};

template <typename AT, int KS>
__device__ __forceinline__ void issue_stage(const AT* __restrict__ A, int lda,
                                            const __bf16* __restrict__ Bt, int ldb,
                                            int m_block, int n_block, int k0,
                                            unsigned sAo, unsigned sBo) {
  using C = StageCfg<AT, KS>;
  const int t = threadIdx.x;
#pragma unroll
  for (int c0 = 0; c0 < C::CA; c0 += 256) {
    const int c   = c0 + t;
    const int row = c / C::CPRA;
    const int cc  = c - row * C::CPRA;
    async_b128(sAo + (unsigned)(row * C::KSPA * (int)sizeof(AT) + cc * 16),
               A + (size_t)(m_block + row) * lda + (k0 + cc * C::EPCA));
  }
#pragma unroll
  for (int c0 = 0; c0 < C::CB; c0 += 256) {
    const int c   = c0 + t;
    const int row = c / C::CPRB;
    const int cc  = c - row * C::CPRB;
    async_b128(sBo + (unsigned)(row * C::KSPB * 2 + cc * 16),
               Bt + (size_t)(n_block + row) * ldb + (k0 + cc * 8));
  }
}

template <typename AT, int KS>
__device__ __forceinline__ void gemm_block_staged(const AT* __restrict__ A, int lda,
                                                  const __bf16* __restrict__ Bt, int ldb,
                                                  int m_block, int n_block, int K,
                                                  v8f acc[4], char* smem) {
  using C = StageCfg<AT, KS>;
  AT*     sA = (AT*)smem;
  __bf16* sB = (__bf16*)(smem + 2 * C::A_BYTES);
  const unsigned sAo = (unsigned)(size_t)(void*)sA;
  const unsigned sBo = (unsigned)(size_t)(void*)sB;
  const int wid = threadIdx.x >> 5, lane = threadIdx.x & 31;
  const int nstages = K / KS;

  issue_stage<AT, KS>(A, lda, Bt, ldb, m_block, n_block, 0, sAo, sBo);
  for (int s = 0; s < nstages; ++s) {
    const int cur = s & 1;
    if (s + 1 < nstages) {
      const int nxt = (s + 1) & 1;
      issue_stage<AT, KS>(A, lda, Bt, ldb, m_block, n_block, (s + 1) * KS,
                          sAo + (unsigned)(nxt * C::A_BYTES),
                          sBo + (unsigned)(nxt * C::B_BYTES));
      wait_async<C::OPS>();
    } else {
      wait_async<0>();
    }
    __syncthreads();
    const AT*     tA = sA + (size_t)cur * (C::A_BYTES / sizeof(AT));
    const __bf16* tB = sB + (size_t)cur * (C::B_BYTES / 2);
#pragma unroll
    for (int kk = 0; kk < KS; kk += 32) {
      v16bf a = lds_frag_f32(tA, C::KSPA, wid * 16, kk, lane);
      v16bf bfr[4];
#pragma unroll
      for (int j = 0; j < 4; ++j)
        bfr[j] = lds_frag_bf16(tB, C::KSPB, j * 16, kk, lane);
#pragma unroll
      for (int j = 0; j < 4; ++j) acc[j] = wmma_bf16(a, bfr[j], acc[j]);
    }
    __syncthreads();
  }
}

// ---------- prep kernels ----------

__global__ void k_cvt_bf16(const float* __restrict__ src, __bf16* __restrict__ dst, int n) {
  int i = blockIdx.x * blockDim.x + threadIdx.x;
  if (i < n) dst[i] = f2bf(src[i]);
}

// src [R x C] f32 row-major -> dst [C x R] bf16 (N-major operand for GEMMs)
__global__ void k_transpose_bf16(const float* __restrict__ src, __bf16* __restrict__ dst,
                                 int R, int C) {
  int i = blockIdx.x * blockDim.x + threadIdx.x;
  if (i >= R * C) return;
  int r = i / C, c = i - r * C;
  dst[(size_t)c * R + r] = f2bf(src[i]);
}

// ---------- GEMM kernels ----------

// Q/K/V projection. mode 0/1: write head-major [B,H,S,HD]; mode 2: write V^T [B,H,HD,S].
__global__ void k_gemm_qkv(const __bf16* __restrict__ xb, const __bf16* __restrict__ Wt,
                           const float* __restrict__ bias, __bf16* __restrict__ out, int mode) {
  __shared__ __align__(16) char smem[SMEM128(32)];
  const int wid = threadIdx.x >> 5, lane = threadIdx.x & 31;
  const int widM = wid & 3, widN = wid >> 2;
  const int m_block = blockIdx.x * 128;
  const int n_block = blockIdx.y * 128;
  v8f acc[2][4]; zero_acc24(acc);
  gemm128x128<32>(xb, D_, Wt, D_, m_block, n_block, D_, acc, smem);
  const int colo = lane & 15, rowo = (lane >> 4) * 8;
#pragma unroll
  for (int mi = 0; mi < 2; ++mi)
#pragma unroll
    for (int j = 0; j < 4; ++j)
#pragma unroll
      for (int rr = 0; rr < 8; ++rr) {
        const int row = m_block + widM * 32 + mi * 16 + rowo + rr;  // token in [0,BS)
        const int col = n_block + widN * 64 + j * 16 + colo;        // feature in [0,D)
        const float v = acc[mi][j][rr] + bias[col];
        const int b = row >> 11, s = row & (S_ - 1);
        const int h = col >> 6,  hd = col & (HD_ - 1);
        size_t idx;
        if (mode < 2) idx = (((size_t)(b * H_ + h) * S_ + s) * HD_ + hd);
        else          idx = (((size_t)(b * H_ + h) * HD_ + hd) * S_ + s);
        out[idx] = f2bf(v);
      }
}

// scores = Q @ K^T per (b,h); raw scores written straight into the atn output region.
__global__ void k_gemm_scores(const __bf16* __restrict__ q, const __bf16* __restrict__ k,
                              float* __restrict__ scores) {
  __shared__ __align__(16) char smem[SMEM128(32)];
  const int bh = blockIdx.z;
  const __bf16* qa = q + (size_t)bh * S_ * HD_;
  const __bf16* kb = k + (size_t)bh * S_ * HD_;   // K rows == Bt (N-major) for Q@K^T
  float* c = scores + (size_t)bh * S_ * S_;
  const int wid = threadIdx.x >> 5, lane = threadIdx.x & 31;
  const int widM = wid & 3, widN = wid >> 2;
  const int m_block = blockIdx.x * 128;
  const int n_block = blockIdx.y * 128;
  v8f acc[2][4]; zero_acc24(acc);
  gemm128x128<32>(qa, HD_, kb, HD_, m_block, n_block, HD_, acc, smem);
  const int colo = lane & 15, rowo = (lane >> 4) * 8;
#pragma unroll
  for (int mi = 0; mi < 2; ++mi)
#pragma unroll
    for (int j = 0; j < 4; ++j)
#pragma unroll
      for (int rr = 0; rr < 8; ++rr) {
        const int row = m_block + widM * 32 + mi * 16 + rowo + rr;
        const int col = n_block + widN * 64 + j * 16 + colo;
        c[(size_t)row * S_ + col] = acc[mi][j][rr];
      }
}

// In-place row softmax over atn (2048 per row), 1/sqrt(HD) folded in.
__global__ void k_softmax(float* __restrict__ atn) {
  __shared__ float red[8];
  float* p = atn + (size_t)blockIdx.x * S_;
  const int t = threadIdx.x, wid = t >> 5, lane = t & 31;
  float v[8];
  float m = -1e30f;
#pragma unroll
  for (int i = 0; i < 8; ++i) { v[i] = p[t + i * 256] * 0.125f; m = fmaxf(m, v[i]); }
#pragma unroll
  for (int o = 16; o > 0; o >>= 1) m = fmaxf(m, __shfl_xor(m, o, 32));
  if (lane == 0) red[wid] = m;
  __syncthreads();
  m = red[0];
#pragma unroll
  for (int i = 1; i < 8; ++i) m = fmaxf(m, red[i]);
  __syncthreads();
  float s = 0.f;
#pragma unroll
  for (int i = 0; i < 8; ++i) { v[i] = __expf(v[i] - m); s += v[i]; }
#pragma unroll
  for (int o = 16; o > 0; o >>= 1) s += __shfl_xor(s, o, 32);
  if (lane == 0) red[wid] = s;
  __syncthreads();
  s = 0.f;
#pragma unroll
  for (int i = 0; i < 8; ++i) s += red[i];
  const float inv = 1.0f / s;
#pragma unroll
  for (int i = 0; i < 8; ++i) p[t + i * 256] = v[i] * inv;
}

// attn_out = atn @ V per (b,h). A is f32 (staged in LDS, converted to bf16 fragments);
// Bt = V^T [HD x S]. Writes attn_out both f32 (residual) and bf16 (FFN operand).
__global__ void k_gemm_av(const float* __restrict__ atn, const __bf16* __restrict__ vt,
                          float* __restrict__ aof, __bf16* __restrict__ aob) {
  __shared__ __align__(16) char smem[StageCfg<float, 32>::SMEM];
  const int bh = blockIdx.z;
  const int b = bh >> 3, h = bh & (H_ - 1);
  const float* a = atn + (size_t)bh * S_ * S_;
  const __bf16* bt = vt + (size_t)bh * HD_ * S_;
  const int wid = threadIdx.x >> 5, lane = threadIdx.x & 31;
  const int m_block = blockIdx.x * 128;
  v8f acc[4];
#pragma unroll
  for (int j = 0; j < 4; ++j)
#pragma unroll
    for (int i = 0; i < 8; ++i) acc[j][i] = 0.0f;
  gemm_block_staged<float, 32>(a, S_, bt, S_, m_block, 0, S_, acc, smem);
  const int m_wave = m_block + wid * 16;
  const int colo = lane & 15, rowo = (lane >> 4) * 8;
#pragma unroll
  for (int j = 0; j < 4; ++j)
#pragma unroll
    for (int rr = 0; rr < 8; ++rr) {
      const int s = m_wave + rowo + rr;
      const int d = h * HD_ + j * 16 + colo;
      const size_t idx = ((size_t)(b * S_ + s)) * D_ + d;
      const float v = acc[j][rr];
      aof[idx] = v;
      aob[idx] = f2bf(v);
    }
}

// h = relu(attn_out @ W1 + b1)
__global__ void k_gemm_ffn1(const __bf16* __restrict__ ao, const __bf16* __restrict__ w1t,
                            const float* __restrict__ b1, __bf16* __restrict__ hbuf) {
  __shared__ __align__(16) char smem[SMEM128(32)];
  const int wid = threadIdx.x >> 5, lane = threadIdx.x & 31;
  const int widM = wid & 3, widN = wid >> 2;
  const int m_block = blockIdx.x * 128;
  const int n_block = blockIdx.y * 128;
  v8f acc[2][4]; zero_acc24(acc);
  gemm128x128<32>(ao, D_, w1t, D_, m_block, n_block, D_, acc, smem);
  const int colo = lane & 15, rowo = (lane >> 4) * 8;
#pragma unroll
  for (int mi = 0; mi < 2; ++mi)
#pragma unroll
    for (int j = 0; j < 4; ++j)
#pragma unroll
      for (int rr = 0; rr < 8; ++rr) {
        const int row = m_block + widM * 32 + mi * 16 + rowo + rr;
        const int col = n_block + widN * 64 + j * 16 + colo;
        float v = acc[mi][j][rr] + b1[col];
        v = v > 0.0f ? v : 0.0f;
        hbuf[(size_t)row * DFF_ + col] = f2bf(v);
      }
}

// y = h @ W2 + b2 + attn_out (residual), f32
__global__ void k_gemm_ffn2(const __bf16* __restrict__ hbuf, const __bf16* __restrict__ w2t,
                            const float* __restrict__ b2, const float* __restrict__ aof,
                            float* __restrict__ y) {
  __shared__ __align__(16) char smem[SMEM128(32)];
  const int wid = threadIdx.x >> 5, lane = threadIdx.x & 31;
  const int widM = wid & 3, widN = wid >> 2;
  const int m_block = blockIdx.x * 128;
  const int n_block = blockIdx.y * 128;
  v8f acc[2][4]; zero_acc24(acc);
  gemm128x128<32>(hbuf, DFF_, w2t, DFF_, m_block, n_block, DFF_, acc, smem);
  const int colo = lane & 15, rowo = (lane >> 4) * 8;
#pragma unroll
  for (int mi = 0; mi < 2; ++mi)
#pragma unroll
    for (int j = 0; j < 4; ++j)
#pragma unroll
      for (int rr = 0; rr < 8; ++rr) {
        const int row = m_block + widM * 32 + mi * 16 + rowo + rr;
        const int col = n_block + widN * 64 + j * 16 + colo;
        const size_t idx = (size_t)row * D_ + col;
        y[idx] = acc[mi][j][rr] + b2[col] + aof[idx];
      }
}

// LayerNorm over D=512 per row
__global__ void k_layernorm(const float* __restrict__ y, const float* __restrict__ gamma,
                            const float* __restrict__ beta, float* __restrict__ out) {
  __shared__ float rs[8], rq[8];
  const float* p = y + (size_t)blockIdx.x * D_;
  float* o = out + (size_t)blockIdx.x * D_;
  const int t = threadIdx.x, wid = t >> 5, lane = t & 31;
  const float a = p[t], b = p[t + 256];
  float s = a + b, q = a * a + b * b;
#pragma unroll
  for (int off = 16; off > 0; off >>= 1) {
    s += __shfl_xor(s, off, 32);
    q += __shfl_xor(q, off, 32);
  }
  if (lane == 0) { rs[wid] = s; rq[wid] = q; }
  __syncthreads();
  s = 0.f; q = 0.f;
#pragma unroll
  for (int i = 0; i < 8; ++i) { s += rs[i]; q += rq[i]; }
  const float mu  = s * (1.0f / D_);
  const float var = q * (1.0f / D_) - mu * mu;
  const float inv = rsqrtf(var + LN_EPS);
  o[t]       = (a - mu) * inv * gamma[t]       + beta[t];
  o[t + 256] = (b - mu) * inv * gamma[t + 256] + beta[t + 256];
}

// ---------- launch ----------

extern "C" void kernel_launch(void* const* d_in, const int* in_sizes, int n_in,
                              void* d_out, int out_size, void* d_ws, size_t ws_size,
                              hipStream_t stream) {
  (void)in_sizes; (void)n_in; (void)out_size; (void)ws_size;
  const float* x     = (const float*)d_in[0];
  const float* Wq    = (const float*)d_in[1];
  const float* bq    = (const float*)d_in[2];
  const float* Wk    = (const float*)d_in[3];
  const float* bk    = (const float*)d_in[4];
  const float* Wv    = (const float*)d_in[5];
  const float* bv    = (const float*)d_in[6];
  const float* W1    = (const float*)d_in[7];
  const float* b1    = (const float*)d_in[8];
  const float* W2    = (const float*)d_in[9];
  const float* b2    = (const float*)d_in[10];
  const float* gamma = (const float*)d_in[11];
  const float* beta  = (const float*)d_in[12];

  char* ws = (char*)d_ws;
  size_t off = 0;
  auto take = [&](size_t bytes) -> char* {
    char* p = ws + off;
    off = (off + bytes + 255) & ~(size_t)255;
    return p;
  };
  __bf16* xb   = (__bf16*)take((size_t)BS_ * D_ * 2);
  __bf16* wqt  = (__bf16*)take((size_t)D_ * D_ * 2);
  __bf16* wkt  = (__bf16*)take((size_t)D_ * D_ * 2);
  __bf16* wvt  = (__bf16*)take((size_t)D_ * D_ * 2);
  __bf16* w1t  = (__bf16*)take((size_t)D_ * DFF_ * 2);
  __bf16* w2t  = (__bf16*)take((size_t)DFF_ * D_ * 2);
  __bf16* qh   = (__bf16*)take((size_t)BS_ * D_ * 2);
  __bf16* kh   = (__bf16*)take((size_t)BS_ * D_ * 2);
  __bf16* vth  = (__bf16*)take((size_t)BS_ * D_ * 2);
  __bf16* aob  = (__bf16*)take((size_t)BS_ * D_ * 2);
  float*  aof  = (float*) take((size_t)BS_ * D_ * 4);
  __bf16* hbuf = (__bf16*)take((size_t)BS_ * DFF_ * 2);
  float*  ybuf = (float*) take((size_t)BS_ * D_ * 4);

  float* outp = (float*)d_out;
  float* atn  = outp + (size_t)BS_ * D_;   // tuple: (out, atn) concatenated flat

  // 1) precision/layout prep
  k_cvt_bf16<<<(BS_ * D_) / 256, 256, 0, stream>>>(x, xb, BS_ * D_);
  k_transpose_bf16<<<(D_ * D_) / 256, 256, 0, stream>>>(Wq, wqt, D_, D_);
  k_transpose_bf16<<<(D_ * D_) / 256, 256, 0, stream>>>(Wk, wkt, D_, D_);
  k_transpose_bf16<<<(D_ * D_) / 256, 256, 0, stream>>>(Wv, wvt, D_, D_);
  k_transpose_bf16<<<(D_ * DFF_) / 256, 256, 0, stream>>>(W1, w1t, D_, DFF_);
  k_transpose_bf16<<<(DFF_ * D_) / 256, 256, 0, stream>>>(W2, w2t, DFF_, D_);

  // 2) QKV projections
  dim3 gqkv(BS_ / 128, D_ / 128, 1);
  k_gemm_qkv<<<gqkv, 256, 0, stream>>>(xb, wqt, bq, qh, 0);
  k_gemm_qkv<<<gqkv, 256, 0, stream>>>(xb, wkt, bk, kh, 1);
  k_gemm_qkv<<<gqkv, 256, 0, stream>>>(xb, wvt, bv, vth, 2);

  // 3) scores -> atn region of d_out, softmax in place
  dim3 gsc(S_ / 128, S_ / 128, B_ * H_);
  k_gemm_scores<<<gsc, 256, 0, stream>>>(qh, kh, atn);
  k_softmax<<<B_ * H_ * S_, 256, 0, stream>>>(atn);

  // 4) attn_out = atn @ V
  dim3 gav(S_ / 128, 1, B_ * H_);
  k_gemm_av<<<gav, 256, 0, stream>>>(atn, vth, aof, aob);

  // 5) FFN + residual + LayerNorm
  dim3 gf1(BS_ / 128, DFF_ / 128, 1);
  k_gemm_ffn1<<<gf1, 256, 0, stream>>>(aob, w1t, b1, hbuf);
  dim3 gf2(BS_ / 128, D_ / 128, 1);
  k_gemm_ffn2<<<gf2, 256, 0, stream>>>(hbuf, w2t, b2, aof, ybuf);
  k_layernorm<<<BS_, 256, 0, stream>>>(ybuf, gamma, beta, outp);
}
